// QuantumFourierNeuralOperator_55972013801699
// MI455X (gfx1250) — matrix operand order repaired and checked by reading, not compiled, not run on hardware
//
#include <hip/hip_runtime.h>

#define NB 16
#define NH 128
#define NW 128
#define NC 64
#define NM 16
#define NPIX (NB*NH*NW)   // 262144
#define NLAY 4

typedef __attribute__((ext_vector_type(16))) _Float16 v16h;
typedef __attribute__((ext_vector_type(8)))  _Float16 v8h;
typedef __attribute__((ext_vector_type(8)))  float    v8f;
typedef __attribute__((ext_vector_type(2)))  float    v2f;

#define WMMA16(A,B,C) __builtin_amdgcn_wmma_f32_16x16x32_f16(false,(A),false,(B),(short)0,(C),false,false)
#define WMMA4(A,B,C)  __builtin_amdgcn_wmma_f32_16x16x4_f32(false,(A),false,(B),(short)0,(C),false,false)

__device__ __forceinline__ float gelu_tanh(float x) {
  float x3 = x * x * x;
  return 0.5f * x * (1.0f + tanhf(0.7978845608028654f * (x + 0.044715f * x3)));
}

// ---- fragment loaders (ISA 7.12.2 wave32 layouts) -------------------------
// A (16x32 f16): lane = M (lane&15); K(j,lane) = (j&7) + (j>=8?16:0) + (lane>=16?8:0)
// src must be pre-offset to (row0, k0); ld in elements.
__device__ __forceinline__ v16h ld_a_h16(const _Float16* src, int ld) {
  int lane = threadIdx.x & 31;
  const _Float16* p = src + (lane & 15) * ld + ((lane >> 4) << 3);
  v8h lo = *(const v8h*)p;
  v8h hi = *(const v8h*)(p + 16);
  v16h r;
  #pragma unroll
  for (int j = 0; j < 8; ++j) { r[j] = lo[j]; r[8 + j] = hi[j]; }
  return r;
}

// Same A layout but source is f32; convert while building fragment.
__device__ __forceinline__ v16h ld_a_f32(const float* src, int ld) {
  int lane = threadIdx.x & 31;
  const float* p = src + (lane & 15) * ld + ((lane >> 4) << 3);
  v16h r;
  #pragma unroll
  for (int j = 0; j < 8; ++j) r[j] = (_Float16)p[j];
  #pragma unroll
  for (int j = 0; j < 8; ++j) r[8 + j] = (_Float16)p[16 + j];
  return r;
}

// B (32x16 f16): lane = N (lane&15); K(j,lane) = (j&7) + (j>=8?16:0) + (lane>=16?8:0)
// Source layout [N][K] row-major (ld elements); src pre-offset to (n0, k0).
__device__ __forceinline__ v16h ld_b_h16(const _Float16* src, int ld) {
  int lane = threadIdx.x & 31;
  const _Float16* p = src + (lane & 15) * ld + ((lane >> 4) << 3);
  v8h lo = *(const v8h*)p;
  v8h hi = *(const v8h*)(p + 16);
  v16h r;
  #pragma unroll
  for (int j = 0; j < 8; ++j) { r[j] = lo[j]; r[8 + j] = hi[j]; }
  return r;
}

// ---- prep kernels ----------------------------------------------------------

__global__ void k_twiddle(float* twc, float* tws) {
  int i = blockIdx.x * 256 + threadIdx.x;
  if (i >= NM * 128) return;
  int kw = i >> 7, w = i & 127;
  int ph = (kw * w) & 127;                  // exact period-128 reduction
  float th = 0.04908738521234052f * (float)ph;  // 2*pi/128
  twc[i] = cosf(th);
  tws[i] = sinf(th);
}

// w_eff[l][m][o][K=n*64+ci] = sum_s w[l,m,n,ci,o,s] / sqrt(8), as f16
__global__ void k_prep_w(const float* wr, const float* wi,
                         _Float16* wefr, _Float16* wefi) {
  int idx = blockIdx.x * 256 + threadIdx.x;
  if (idx >= NLAY * 16 * 64 * 1024) return;
  int K = idx & 1023;
  int o = (idx >> 10) & 63;
  int m = (idx >> 16) & 15;
  int l = idx >> 20;
  int n = K >> 6, ci = K & 63;
  long src = (((((long)l * 16 + m) * 16 + n) * 64 + ci) * 64 + o) * 8;
  float sr = 0.f, si = 0.f;
  #pragma unroll
  for (int s = 0; s < 8; ++s) { sr += wr[src + s]; si += wi[src + s]; }
  const float sc = 0.35355339059327373f;  // 1/sqrt(8)
  wefr[idx] = (_Float16)(sr * sc);
  wefi[idx] = (_Float16)(si * sc);
}

// wwT16[l][o][ci] and fc1T16[o][ci] (transposed, f16) for WMMA B-fragments
__global__ void k_prep_small(const float* ww, const float* fc1w,
                             _Float16* wwT, _Float16* fc1T) {
  int idx = blockIdx.x * 256 + threadIdx.x;
  if (idx < NLAY * 64 * 64) {
    int l = idx >> 12; int rem = idx & 4095;
    int o = rem >> 6, ci = rem & 63;
    wwT[idx] = (_Float16)ww[(l * 64 + ci) * 64 + o];
  } else if (idx < NLAY * 64 * 64 + 128 * 64) {
    int j = idx - NLAY * 64 * 64;
    int o = j >> 6, ci = j & 63;
    fc1T[j] = (_Float16)fc1w[ci * 128 + o];
  }
}

// ---- lift: h = x @ fc_in_w + b (K=3, pure VALU, bandwidth bound) ----------
__global__ void k_lift(const float* x, const float* w, const float* b, float* h) {
  int idx = blockIdx.x * 256 + threadIdx.x;   // NPIX*NC threads
  int p = idx >> 6, c = idx & 63;
  float acc = b[c];
  #pragma unroll
  for (int k = 0; k < 3; ++k) acc += x[p * 3 + k] * w[k * 64 + c];
  h[idx] = acc;
}

// ---- forward DFT stage A (contract W): tmpA[b,h,kw,c] -----------------
// f32 WMMA: A = 16x128 twiddles, B = 128x64 slab of h. 1 block per (b,h).
__global__ void k_dftA(const float* hbuf, const float* twc, const float* tws,
                       float* tAr, float* tAi) {
  int slab = blockIdx.x;                 // b*NH + hrow
  int wave = threadIdx.x >> 5;           // c tile
  int lane = threadIdx.x & 31;
  const float* src = hbuf + (size_t)slab * (NW * NC);
  int mrow = lane & 15;                  // kw
  int koff = (lane >> 4) << 1;
  int c = wave * 16 + (lane & 15);
  v8f accr = {}, acci = {};
  for (int k0 = 0; k0 < 128; k0 += 4) {
    int k = k0 + koff;
    v2f ac = { twc[mrow * 128 + k], twc[mrow * 128 + k + 1] };
    v2f as = { tws[mrow * 128 + k], tws[mrow * 128 + k + 1] };
    v2f bf = { src[k * NC + c], src[(k + 1) * NC + c] };
    accr = WMMA4(ac, bf, accr);
    acci = WMMA4(as, bf, acci);
  }
  #pragma unroll
  for (int r = 0; r < 8; ++r) {
    int kw = r + ((lane >> 4) << 3);
    int idx = (slab * NM + kw) * NC + c;
    tAr[idx] = accr[r];
    tAi[idx] = -acci[r];                 // e^{-i theta}: imag = -sin
  }
}

// ---- forward DFT stage B (contract H): x_ft (stored directly as f16) ------
// 1 block per (b,kw); complex x complex.
__global__ void k_dftB(const float* tAr, const float* tAi,
                       const float* twc, const float* tws,
                       _Float16* xftr, _Float16* xfti) {
  int slab = blockIdx.x;                 // b*16 + kw
  int b = slab >> 4, kw = slab & 15;
  int wave = threadIdx.x >> 5;
  int lane = threadIdx.x & 31;
  int c = wave * 16 + (lane & 15);
  int khm = lane & 15;                   // M = kh
  int koff = (lane >> 4) << 1;
  const float* brb = tAr + (size_t)b * (NH * NM * NC) + kw * NC + c;
  const float* bib = tAi + (size_t)b * (NH * NM * NC) + kw * NC + c;
  v8f accr = {}, acci = {};
  for (int k0 = 0; k0 < 128; k0 += 4) {
    int k = k0 + koff;
    v2f ac = { twc[khm * 128 + k], twc[khm * 128 + k + 1] };
    float s0 = tws[khm * 128 + k], s1 = tws[khm * 128 + k + 1];
    v2f as  = {  s0,  s1 };
    v2f asn = { -s0, -s1 };
    v2f br = { brb[(size_t)k * 1024], brb[(size_t)(k + 1) * 1024] };
    v2f bi = { bib[(size_t)k * 1024], bib[(size_t)(k + 1) * 1024] };
    accr = WMMA4(ac, br, accr);    // (c - i s)(tr + i ti)
    accr = WMMA4(as, bi, accr);
    acci = WMMA4(ac, bi, acci);
    acci = WMMA4(asn, br, acci);
  }
  #pragma unroll
  for (int r = 0; r < 8; ++r) {
    int kh = r + ((lane >> 4) << 3);
    int idx = (b * 16 + kh) * 1024 + kw * 64 + c;  // [row=(b,kh)][K=(kw,c)]
    xftr[idx] = (_Float16)accr[r];
    xfti[idx] = (_Float16)acci[r];
  }
}

// ---- spectral contraction: 16 complex GEMMs M=256 N=64 K=1024, f16 WMMA ---
__global__ void k_einsum(const _Float16* xftr, const _Float16* xfti,
                         const _Float16* wefr, const _Float16* wefi,
                         float* oftr, float* ofti) {
  int m = blockIdx.x;                    // output mode
  int rt = blockIdx.y;                   // row tile of 16 rows
  int wave = threadIdx.x >> 5;
  int lane = threadIdx.x & 31;
  int n0 = wave * 16;
  const _Float16* arb = xftr + rt * 16 * 1024;
  const _Float16* aib = xfti + rt * 16 * 1024;
  const _Float16* brb = wefr + ((size_t)m * 64 + n0) * 1024;
  const _Float16* bib = wefi + ((size_t)m * 64 + n0) * 1024;
  v8f accr = {}, acci = {};
  for (int k0 = 0; k0 < 1024; k0 += 32) {
    v16h Ar = ld_a_h16(arb + k0, 1024);
    v16h Ai = ld_a_h16(aib + k0, 1024);
    v16h An;                             // -Ai (ISA: NEG on A/B illegal for fp)
    #pragma unroll
    for (int j = 0; j < 16; ++j) An[j] = -Ai[j];
    v16h Br = ld_b_h16(brb + k0, 1024);
    v16h Bi = ld_b_h16(bib + k0, 1024);
    accr = WMMA16(Ar, Br, accr);
    accr = WMMA16(An, Bi, accr);
    acci = WMMA16(Ar, Bi, acci);
    acci = WMMA16(Ai, Br, acci);
  }
  #pragma unroll
  for (int r = 0; r < 8; ++r) {
    int gr = rt * 16 + r + ((lane >> 4) << 3);      // (b, X)
    int idx = (gr * 16 + m) * 64 + n0 + (lane & 15);
    oftr[idx] = accr[r];
    ofti[idx] = acci[r];
  }
}

// ---- pointwise dense: x2 = h @ ww + wb, f16 WMMA, K=64 --------------------
__global__ void k_point(const float* hbuf, const _Float16* wwT,
                        const float* wb, float* x2) {
  int rt = blockIdx.x;                   // 16-row tile of NPIX
  int wave = threadIdx.x >> 5;
  int lane = threadIdx.x & 31;
  int n0 = wave * 16;
  v8f acc;
  float bias = wb[n0 + (lane & 15)];
  #pragma unroll
  for (int r = 0; r < 8; ++r) acc[r] = bias;
  #pragma unroll
  for (int k0 = 0; k0 < 64; k0 += 32) {
    v16h a = ld_a_f32(hbuf + (size_t)rt * 16 * NC + k0, NC);
    v16h b = ld_b_h16(wwT + n0 * 64 + k0, 64);
    acc = WMMA16(a, b, acc);
  }
  #pragma unroll
  for (int r = 0; r < 8; ++r) {
    int row = rt * 16 + r + ((lane >> 4) << 3);
    x2[(size_t)row * NC + n0 + (lane & 15)] = acc[r];
  }
}

// ---- inverse DFT stage 1 (expand kh -> hh): t2[b,hh,kw,c] -----------------
__global__ void k_inv1(const float* oftr, const float* ofti,
                       const float* twc, const float* tws,
                       float* t2r, float* t2i) {
  int slab = blockIdx.x;                 // b*16 + kw
  int ntile = blockIdx.y;
  int mtile = threadIdx.x >> 5;          // hh tile (0..7)
  int lane = threadIdx.x & 31;
  int b = slab >> 4, kw = slab & 15;
  int hh = mtile * 16 + (lane & 15);
  int koff = (lane >> 4) << 1;
  int c = ntile * 16 + (lane & 15);
  const float* brb = oftr + (size_t)b * 16384 + kw * 64 + c;  // + kh*1024
  const float* bib = ofti + (size_t)b * 16384 + kw * 64 + c;
  v8f ar = {}, ai = {};
  #pragma unroll
  for (int k0 = 0; k0 < 16; k0 += 4) {
    int k = k0 + koff;
    v2f ac = { twc[k * 128 + hh], twc[(k + 1) * 128 + hh] };
    float s0 = tws[k * 128 + hh], s1 = tws[(k + 1) * 128 + hh];
    v2f as  = {  s0,  s1 };
    v2f asn = { -s0, -s1 };
    v2f br = { brb[k * 1024], brb[(k + 1) * 1024] };
    v2f bi = { bib[k * 1024], bib[(k + 1) * 1024] };
    ar = WMMA4(ac, br, ar);              // (c + i s)(or + i oi)
    ar = WMMA4(asn, bi, ar);
    ai = WMMA4(ac, bi, ai);
    ai = WMMA4(as, br, ai);
  }
  #pragma unroll
  for (int r = 0; r < 8; ++r) {
    int hhr = mtile * 16 + r + ((lane >> 4) << 3);
    int idx = ((b * NH + hhr) * NM + kw) * NC + c;
    t2r[idx] = ar[r];
    t2i[idx] = ai[r];
  }
}

// ---- inverse DFT stage 2 (expand kw -> ww), fused + x2 + gelu -> h --------
__global__ void k_inv2(const float* t2r, const float* t2i, const float* x2,
                       const float* twc, const float* tws, float* hbuf) {
  int slab = blockIdx.x;                 // b*NH + hh
  int ntile = blockIdx.y;
  int mtile = threadIdx.x >> 5;          // ww tile
  int lane = threadIdx.x & 31;
  int wwc = mtile * 16 + (lane & 15);
  int koff = (lane >> 4) << 1;
  int c = ntile * 16 + (lane & 15);
  const float* brb = t2r + (size_t)slab * 1024 + c;   // + kw*64
  const float* bib = t2i + (size_t)slab * 1024 + c;
  v8f acc = {};
  #pragma unroll
  for (int k0 = 0; k0 < 16; k0 += 4) {
    int k = k0 + koff;
    v2f ac  = { twc[k * 128 + wwc], twc[(k + 1) * 128 + wwc] };
    v2f asn = { -tws[k * 128 + wwc], -tws[(k + 1) * 128 + wwc] };
    v2f br = { brb[k * 64], brb[(k + 1) * 64] };
    v2f bi = { bib[k * 64], bib[(k + 1) * 64] };
    acc = WMMA4(ac, br, acc);            // Re((c + i s)(tr + i ti)) = c*tr - s*ti
    acc = WMMA4(asn, bi, acc);
  }
  const float inv = 1.0f / 16384.0f;     // ifftn normalization
  #pragma unroll
  for (int r = 0; r < 8; ++r) {
    int w = mtile * 16 + r + ((lane >> 4) << 3);
    size_t idx = ((size_t)slab * NW + w) * NC + c;
    float v = acc[r] * inv + x2[idx];
    hbuf[idx] = gelu_tanh(v);
  }
}

// ---- head: out = gelu(h @ fc1 + b1) @ fc2 + b2, one wave per 16 rows ------
__global__ void k_head(const float* hbuf, const _Float16* fc1T,
                       const float* fc1b, const float* fc2w, const float* fc2b,
                       float* out) {
  int rt = blockIdx.x;
  int lane = threadIdx.x & 31;
  const float* arow = hbuf + (size_t)rt * 16 * NC;
  v16h a0 = ld_a_f32(arow + 0, NC);
  v16h a1 = ld_a_f32(arow + 32, NC);
  float part[8];
  #pragma unroll
  for (int r = 0; r < 8; ++r) part[r] = 0.f;
  #pragma unroll
  for (int t = 0; t < 8; ++t) {
    int n = t * 16 + (lane & 15);
    float bias = fc1b[n];
    v8f acc;
    #pragma unroll
    for (int r = 0; r < 8; ++r) acc[r] = bias;
    v16h b0 = ld_b_h16(fc1T + (t * 16) * 64 + 0, 64);
    v16h b1 = ld_b_h16(fc1T + (t * 16) * 64 + 32, 64);
    acc = WMMA16(a0, b0, acc);
    acc = WMMA16(a1, b1, acc);
    float f2 = fc2w[n];
    #pragma unroll
    for (int r = 0; r < 8; ++r) part[r] += gelu_tanh(acc[r]) * f2;
  }
  #pragma unroll
  for (int m = 1; m < 16; m <<= 1) {
    #pragma unroll
    for (int r = 0; r < 8; ++r) part[r] += __shfl_xor(part[r], m, 32);
  }
  if ((lane & 15) == 0) {
    float b2 = fc2b[0];
    int rb = rt * 16 + ((lane >> 4) << 3);
    #pragma unroll
    for (int r = 0; r < 8; ++r) out[rb + r] = part[r] + b2;
  }
}

// ---- host side -------------------------------------------------------------
extern "C" void kernel_launch(void* const* d_in, const int* in_sizes, int n_in,
                              void* d_out, int out_size, void* d_ws, size_t ws_size,
                              hipStream_t stream) {
  (void)in_sizes; (void)n_in; (void)out_size; (void)ws_size;
  const float* x       = (const float*)d_in[0];
  const float* fc_in_w = (const float*)d_in[1];
  const float* fc_in_b = (const float*)d_in[2];
  const float* wr      = (const float*)d_in[3];
  const float* wi      = (const float*)d_in[4];
  const float* ww      = (const float*)d_in[5];
  const float* wb      = (const float*)d_in[6];
  const float* fc1_w   = (const float*)d_in[7];
  const float* fc1_b   = (const float*)d_in[8];
  const float* fc2_w   = (const float*)d_in[9];
  const float* fc2_b   = (const float*)d_in[10];

  char* base = (char*)d_ws;
  size_t off = 0;
  auto carve = [&](size_t bytes) -> void* {
    void* p = base + off;
    off = (off + bytes + 255) & ~(size_t)255;
    return p;
  };
  float*    hbuf = (float*)carve((size_t)NPIX * NC * 4);           // 67 MB
  float*    x2   = (float*)carve((size_t)NPIX * NC * 4);           // 67 MB
  float*    tAr  = (float*)carve((size_t)NB * NH * NM * NC * 4);   // 8 MB
  float*    tAi  = (float*)carve((size_t)NB * NH * NM * NC * 4);
  float*    t2r  = (float*)carve((size_t)NB * NH * NM * NC * 4);
  float*    t2i  = (float*)carve((size_t)NB * NH * NM * NC * 4);
  _Float16* xftr = (_Float16*)carve((size_t)256 * 1024 * 2);
  _Float16* xfti = (_Float16*)carve((size_t)256 * 1024 * 2);
  float*    oftr = (float*)carve((size_t)256 * 16 * 64 * 4);
  float*    ofti = (float*)carve((size_t)256 * 16 * 64 * 4);
  _Float16* wefr = (_Float16*)carve((size_t)NLAY * 16 * 64 * 1024 * 2);  // 8 MB
  _Float16* wefi = (_Float16*)carve((size_t)NLAY * 16 * 64 * 1024 * 2);
  _Float16* wwT  = (_Float16*)carve((size_t)NLAY * 64 * 64 * 2);
  _Float16* fc1T = (_Float16*)carve((size_t)128 * 64 * 2);
  float*    twc  = (float*)carve((size_t)NM * 128 * 4);
  float*    tws  = (float*)carve((size_t)NM * 128 * 4);

  k_twiddle<<<8, 256, 0, stream>>>(twc, tws);
  k_prep_w<<<(NLAY * 16 * 64 * 1024) / 256, 256, 0, stream>>>(wr, wi, wefr, wefi);
  k_prep_small<<<96, 256, 0, stream>>>(ww, fc1_w, wwT, fc1T);
  k_lift<<<(NPIX * NC) / 256, 256, 0, stream>>>(x, fc_in_w, fc_in_b, hbuf);

  for (int l = 0; l < NLAY; ++l) {
    k_dftA<<<NB * NH, 128, 0, stream>>>(hbuf, twc, tws, tAr, tAi);
    k_dftB<<<256, 128, 0, stream>>>(tAr, tAi, twc, tws, xftr, xfti);
    k_einsum<<<dim3(16, 16, 1), 128, 0, stream>>>(
        xftr, xfti, wefr + (size_t)l * 16 * 64 * 1024,
        wefi + (size_t)l * 16 * 64 * 1024, oftr, ofti);
    k_point<<<NPIX / 16, 128, 0, stream>>>(hbuf, wwT + l * 4096, wb + l * 64, x2);
    k_inv1<<<dim3(256, 4, 1), 256, 0, stream>>>(oftr, ofti, twc, tws, t2r, t2i);
    k_inv2<<<dim3(NB * NH, 4, 1), 256, 0, stream>>>(t2r, t2i, x2, twc, tws, hbuf);
  }

  k_head<<<NPIX / 16, 32, 0, stream>>>(hbuf, fc1T, fc1_b, fc2_w, fc2_b, (float*)d_out);
}
